// GateL0RDModel_84018150245163
// MI455X (gfx1250) — matrix operand order repaired and checked by reading, not compiled, not run on hardware
//
#include <hip/hip_runtime.h>
#include <hip/hip_bf16.h>

typedef _Float16 v16h __attribute__((ext_vector_type(16)));
typedef _Float16 v8h  __attribute__((ext_vector_type(8)));
typedef float    v8f  __attribute__((ext_vector_type(8)));

#define T_STEPS 100
#define BATCH   256
#define DOBS    64
#define DFEAT   512
#define DLAT    512
#define DOUTSZ  64
// Flat output offsets (floats): outs, hs, thetas, ys, delta_outs
#define OFF_OUTS  0
#define OFF_HS    1638400
#define OFF_TH    14745600
#define OFF_YS    27852800
#define OFF_DO    40960000

// ---------------------------------------------------------------------------
// Weight pre-pack: f32 (K x N row-major) -> f16 WMMA B-fragment layout.
// Fragment (kc, nt) = 32 lanes x 16 halves, per CDNA5 16-bit B layout:
//   lane l<16 : N = nt*16+l, dword j holds K = kc*32 + (j<4 ? 2j : 16+2(j-4)), +1
//   lane l>=16: N = nt*16+(l-16), dword j holds K = kc*32 + 8 + (j<4 ? 2j : 16+2(j-4)), +1
// ---------------------------------------------------------------------------
__global__ __launch_bounds__(256)
void pack_weights_f16(const float* __restrict__ src, _Float16* __restrict__ dst,
                      int K, int N) {
    int d = blockIdx.x * blockDim.x + threadIdx.x;     // one dword (2 halves)
    int total = (K * N) >> 1;
    if (d >= total) return;
    int frag   = d >> 8;           // 256 dwords per fragment
    int within = d & 255;
    int lane   = within >> 3;
    int j      = within & 7;
    int ntiles = N >> 4;
    int kc     = frag / ntiles;
    int nt     = frag - kc * ntiles;
    int col    = nt * 16 + (lane & 15);
    int k0     = kc * 32 + ((lane & 16) ? 8 : 0) + ((j & 4) ? 16 : 0) + 2 * (j & 3);
    dst[2 * d]     = (_Float16)src[k0 * N + col];
    dst[2 * d + 1] = (_Float16)src[(k0 + 1) * N + col];
}

// ---------------------------------------------------------------------------
// A-fragment gather from an f16 row-major activation tile in LDS.
// CDNA5 16-bit A layout: lane l<16 = row l, halves {k0..k0+7, k0+16..k0+23};
// lane l>=16 = row l-16, offset +8.
// ---------------------------------------------------------------------------
__device__ __forceinline__ v16h load_afrag(const _Float16* base, int lane,
                                           int rowStride, int kc) {
    int row = lane & 15;
    int k0  = kc + ((lane & 16) ? 8 : 0);
    const _Float16* p = base + row * rowStride + k0;
    v8h lo = *(const v8h*)(p);
    v8h hi = *(const v8h*)(p + 16);
    return __builtin_shufflevector(lo, hi, 0, 1, 2, 3, 4, 5, 6, 7,
                                   8, 9, 10, 11, 12, 13, 14, 15);
}

__device__ __forceinline__ v16h load_bfrag(const _Float16* wpack, int lane,
                                           int ntiles, int kcIdx, int nt) {
    const _Float16* p = wpack + ((size_t)((kcIdx * ntiles + nt) * 32 + lane) << 4);
    return *(const v16h*)(p);     // 32B contiguous per lane
}

__device__ __forceinline__ v8f wmma_f16(v16h a, v16h b, v8f c) {
    return __builtin_amdgcn_wmma_f32_16x16x32_f16(false, a, false, b,
                                                  (short)0, c, false, false);
}

// Branch-free tanh on the native transcendental unit:
// tanh(x) = sign(x) * (1 - e) / (1 + e), e = exp(-2|x|).
// For large |x|, e -> 0 and the expression saturates to +/-1 cleanly.
__device__ __forceinline__ float tanh_fast(float x) {
    float ax = __builtin_fabsf(x);
    float e  = __expf(-2.0f * ax);                       // v_exp_f32
    float r  = (1.0f - e) * __builtin_amdgcn_rcpf(1.0f + e);  // v_rcp_f32
    return __builtin_copysignf(r, x);
}

__device__ __forceinline__ float sigmoid_fast(float x) {
    return __builtin_amdgcn_rcpf(1.0f + __expf(-x));     // v_exp + v_rcp
}

// ---------------------------------------------------------------------------
// Persistent recurrent kernel. grid = 16 WGs x 256 threads (8 waves).
// Each WG owns a 16-row batch tile for the entire T=100 scan (no inter-WG
// deps: h/out recurrences are per-row). Wave w owns N-tiles 4w..4w+3.
// ---------------------------------------------------------------------------
__global__ __launch_bounds__(256)
void gatel0rd_scan_kernel(const float* __restrict__ obs,
                          const float* __restrict__ smask,
                          const float* __restrict__ noise,
                          const float* __restrict__ b_pre,
                          const float* __restrict__ bg,
                          const float* __restrict__ br,
                          const float* __restrict__ bp,
                          const float* __restrict__ bo,
                          const float* __restrict__ b_out,
                          const _Float16* __restrict__ wpre,
                          const _Float16* __restrict__ wg,
                          const _Float16* __restrict__ wr,
                          const _Float16* __restrict__ wp_,
                          const _Float16* __restrict__ wo,
                          const _Float16* __restrict__ wout,
                          float* __restrict__ out) {
    extern __shared__ char smem[];
    _Float16* s_gr   = (_Float16*)smem;                 // 16 x 1024 f16: [x | h_prev] -> [x | h] -> y in cols 512..1023
    float*    s_h    = (float*)(smem + 16 * 1024 * 2);  // 16 x 512 f32: h state
    float*    s_xin  = (float*)(smem + 16 * 1024 * 2 + 16 * 512 * 4); // 16 x 64 f32
    float*    s_outp = s_xin + 16 * 64;                 // 16 x 64 f32: last_out
    _Float16* s_xinh = (_Float16*)(s_outp + 16 * 64);   // 16 x 64 f16

    const int tid  = threadIdx.x;
    const int lane = tid & 31;
    const int wave = tid >> 5;
    const int r0   = blockIdx.x * 16;    // batch-row tile base

    // init state: h = 0, last_out = 0, gr h-half = 0
    for (int i = tid; i < 16 * 512; i += 256) s_h[i] = 0.0f;
    for (int i = tid; i < 16 * 64;  i += 256) s_outp[i] = 0.0f;
    for (int i = tid; i < 16 * 512; i += 256)
        s_gr[(i >> 9) * 1024 + 512 + (i & 511)] = (_Float16)0.0f;
    __syncthreads();

    float* outs_o = out + OFF_OUTS;
    float* hs_o   = out + OFF_HS;
    float* th_o   = out + OFF_TH;
    float* ys_o   = out + OFF_YS;
    float* do_o   = out + OFF_DO;

    const int m0 = (lane & 16) ? 8 : 0;
    const int nl = lane & 15;

    for (int t = 0; t < T_STEPS; ++t) {
        // ---- x_in = m*obs + (1-m)*last_out -----------------------------
        for (int e = tid; e < 16 * 64; e += 256) {
            int m = e >> 6, d = e & 63;
            int r = r0 + m;
            float mv = (t == 0) ? 1.0f
                                : ((smask[t * BATCH + r] < 0.5f) ? 1.0f : 0.0f);
            float xi = mv * obs[(size_t)(t * BATCH + r) * DOBS + d]
                     + (1.0f - mv) * s_outp[e];
            s_xin[e]  = xi;
            s_xinh[e] = (_Float16)xi;
        }
        __syncthreads();

        // ---- GEMM1: x = tanh(x_in @ W_pre + b_pre) -> gr[:, 0:512] -----
        {
            v8f acc[4] = {};
            for (int kc = 0; kc < 2; ++kc) {
                v16h a = load_afrag(s_xinh, lane, DOBS, kc * 32);
#pragma unroll
                for (int i = 0; i < 4; ++i) {
                    v16h b = load_bfrag(wpre, lane, 32, kc, wave * 4 + i);
                    acc[i] = wmma_f16(a, b, acc[i]);
                }
            }
#pragma unroll
            for (int i = 0; i < 4; ++i) {
                int nt = wave * 4 + i;
#pragma unroll
                for (int v = 0; v < 8; ++v) {
                    int row = v + m0;
                    int n   = nt * 16 + nl;
                    float x = tanh_fast(acc[i][v] + b_pre[n]);
                    s_gr[row * 1024 + n] = (_Float16)x;
                }
            }
        }
        __syncthreads();

        // ---- GEMM2: g/R over gr=[x,h_prev], K=1024; shared A-frags -----
        {
            v8f cg[4] = {}, cr[4] = {};
            for (int kc = 0; kc < 32; ++kc) {
                v16h a = load_afrag(s_gr, lane, 1024, kc * 32);
                if (kc + 1 < 32) {   // prefetch next chunk's fragments into L2/L0
                    __builtin_prefetch(wg + ((size_t)(((kc + 1) * 32 + wave * 4) * 32 + lane) << 4), 0, 1);
                    __builtin_prefetch(wr + ((size_t)(((kc + 1) * 32 + wave * 4) * 32 + lane) << 4), 0, 1);
                }
#pragma unroll
                for (int i = 0; i < 4; ++i) {
                    int nt = wave * 4 + i;
                    cg[i] = wmma_f16(a, load_bfrag(wg, lane, 32, kc, nt), cg[i]);
                    cr[i] = wmma_f16(a, load_bfrag(wr, lane, 32, kc, nt), cr[i]);
                }
            }
            __syncthreads();   // all K-loop reads of gr h-half complete
#pragma unroll
            for (int i = 0; i < 4; ++i) {
                int nt = wave * 4 + i;
#pragma unroll
                for (int v = 0; v < 8; ++v) {
                    int row = v + m0;
                    int n   = nt * 16 + nl;
                    size_t gi = (size_t)(t * BATCH + r0 + row) * DLAT + n;
                    float g = tanh_fast(cg[i][v] + bg[n] + noise[gi]);
                    g = fmaxf(g, 0.0f);
                    float theta = (g > 0.0f) ? 1.0f : 0.0f;
                    float R  = tanh_fast(cr[i][v] + br[n]);
                    float hp = s_h[row * 512 + n];
                    float h  = g * R + (1.0f - g) * hp;
                    s_h[row * 512 + n] = h;
                    th_o[gi] = theta;
                    hs_o[gi] = h;
                    s_gr[row * 1024 + 512 + n] = (_Float16)h;  // p_in h-half
                }
            }
        }
        __syncthreads();

        // ---- GEMM3: y = tanh(p_in@Wp+bp) * sigmoid(p_in@Wo+bo) ---------
        {
            v8f cp[4] = {}, co[4] = {};
            for (int kc = 0; kc < 32; ++kc) {
                v16h a = load_afrag(s_gr, lane, 1024, kc * 32);
                if (kc + 1 < 32) {
                    __builtin_prefetch(wp_ + ((size_t)(((kc + 1) * 32 + wave * 4) * 32 + lane) << 4), 0, 1);
                    __builtin_prefetch(wo  + ((size_t)(((kc + 1) * 32 + wave * 4) * 32 + lane) << 4), 0, 1);
                }
#pragma unroll
                for (int i = 0; i < 4; ++i) {
                    int nt = wave * 4 + i;
                    cp[i] = wmma_f16(a, load_bfrag(wp_, lane, 32, kc, nt), cp[i]);
                    co[i] = wmma_f16(a, load_bfrag(wo, lane, 32, kc, nt), co[i]);
                }
            }
            __syncthreads();   // all K-loop reads of p_in complete
#pragma unroll
            for (int i = 0; i < 4; ++i) {
                int nt = wave * 4 + i;
#pragma unroll
                for (int v = 0; v < 8; ++v) {
                    int row = v + m0;
                    int n   = nt * 16 + nl;
                    float P = tanh_fast(cp[i][v] + bp[n]);
                    float O = sigmoid_fast(co[i][v] + bo[n]);
                    float y = P * O;
                    ys_o[(size_t)(t * BATCH + r0 + row) * DFEAT + n] = y;
                    s_gr[row * 1024 + 512 + n] = (_Float16)y;  // y tile for GEMM4
                }
            }
        }
        __syncthreads();

        // ---- GEMM4: out_pre = y @ W_out + b_out; out = x_in + out_pre --
        if (wave < 4) {        // 4 N-tiles of 16 cols; uniform per-wave branch
            v8f acc = {};
            for (int kc = 0; kc < 16; ++kc) {
                v16h a = load_afrag(s_gr + 512, lane, 1024, kc * 32);
                v16h b = load_bfrag(wout, lane, 4, kc, wave);
                acc = wmma_f16(a, b, acc);
            }
#pragma unroll
            for (int v = 0; v < 8; ++v) {
                int row = v + m0;
                int n   = wave * 16 + nl;
                float op = acc[v] + b_out[n];
                float o  = s_xin[row * 64 + n] + op;   // FACTOR_DELTA = 1.0
                size_t oi = (size_t)(t * BATCH + r0 + row) * DOUTSZ + n;
                do_o[oi]   = op;
                outs_o[oi] = o;
                s_outp[row * 64 + n] = o;
            }
        }
        __syncthreads();       // last_out visible for next step's x_in
    }
}

// ---------------------------------------------------------------------------
extern "C" void kernel_launch(void* const* d_in, const int* in_sizes, int n_in,
                              void* d_out, int out_size, void* d_ws, size_t ws_size,
                              hipStream_t stream) {
    (void)in_sizes; (void)n_in; (void)out_size; (void)ws_size;
    const float* obs   = (const float*)d_in[0];
    const float* smask = (const float*)d_in[1];
    const float* noise = (const float*)d_in[2];
    const float* Wpre  = (const float*)d_in[3];
    const float* bpre  = (const float*)d_in[4];
    const float* Wg    = (const float*)d_in[5];
    const float* bg    = (const float*)d_in[6];
    const float* Wr    = (const float*)d_in[7];
    const float* br    = (const float*)d_in[8];
    const float* Wp    = (const float*)d_in[9];
    const float* bp    = (const float*)d_in[10];
    const float* Wo    = (const float*)d_in[11];
    const float* bo    = (const float*)d_in[12];
    const float* Wout  = (const float*)d_in[13];
    const float* bout  = (const float*)d_in[14];

    // f16 packed-weight arena in workspace (~4.3 MB)
    _Float16* ws    = (_Float16*)d_ws;
    _Float16* pWpre = ws;                              // 64*512
    _Float16* pWg   = ws + 32768;                      // 1024*512
    _Float16* pWr   = ws + 32768 + 1 * 524288;
    _Float16* pWp   = ws + 32768 + 2 * 524288;
    _Float16* pWo   = ws + 32768 + 3 * 524288;
    _Float16* pWout = ws + 32768 + 4 * 524288;         // 512*64

    auto pack = [&](const float* s, _Float16* dst, int K, int N) {
        int total = (K * N) / 2;
        pack_weights_f16<<<(total + 255) / 256, 256, 0, stream>>>(s, dst, K, N);
    };
    pack(Wpre, pWpre, 64, 512);
    pack(Wg,   pWg,   1024, 512);
    pack(Wr,   pWr,   1024, 512);
    pack(Wp,   pWp,   1024, 512);
    pack(Wo,   pWo,   1024, 512);
    pack(Wout, pWout, 512, 64);

    // LDS: gr(32KB) + h(32KB) + x_in f32(4KB) + last_out(4KB) + x_in f16(2KB)
    size_t lds = 16 * 1024 * 2 + 16 * 512 * 4 + 16 * 64 * 4 * 2 + 16 * 64 * 2;
    gatel0rd_scan_kernel<<<BATCH / 16, 256, lds, stream>>>(
        obs, smask, noise, bpre, bg, br, bp, bo, bout,
        pWpre, pWg, pWr, pWp, pWo, pWout, (float*)d_out);
}
